// BaselineLSTM_6184752906302
// MI455X (gfx1250) — matrix operand order repaired and checked by reading, not compiled, not run on hardware
//
#include <hip/hip_runtime.h>
#include <hip/hip_bf16.h>

// ---------------------------------------------------------------------------
// 5-layer LSTM for MI455X (gfx1250).
// One persistent kernel launch per layer; 48 workgroups, each owning a
// 16-wide hidden slice. All weights for the slice are converted to bf16 and
// held in LDS (~300KB, fits CDNA5's 320KB/WGP) for the entire T=1024 scan.
// Per timestep each of the 8 waves computes one 16x16 gate tile with
// v_wmma_f32_16x16x32_bf16 (fp32 accumulate); h_{t-1} is staged into LDS with
// GLOBAL_LOAD_ASYNC_TO_LDS_B128 (ASYNCcnt), x_{t+1} is prefetched with
// global_prefetch_b8, and the grid synchronizes with an atomic
// sense-reversing barrier each step.
// ---------------------------------------------------------------------------

typedef __attribute__((ext_vector_type(16))) __bf16 v16bf;
typedef __attribute__((ext_vector_type(8)))  __bf16 v8bf;
typedef __attribute__((ext_vector_type(8)))  float  v8f;

#define HLSTM   768
#define BATCH   32
#define TSTEPS  1024
#define NWG     48       // 768 / 16 hidden slices
#define THREADS 256      // 8 wave32

// ---- helpers ---------------------------------------------------------------

__device__ __forceinline__ __bf16 to_bf16(float f) {
  unsigned u = __float_as_uint(f);
  unsigned r = (u + 0x7FFFu + ((u >> 16) & 1u)) >> 16;   // round-to-nearest-even
  unsigned short s = (unsigned short)r;
  __bf16 b;
  __builtin_memcpy(&b, &s, 2);
  return b;
}

__device__ __forceinline__ float sigmoidf_(float x) {
  return 1.f / (1.f + __expf(-x));
}

// A-matrix (16x32, bf16) lane packing (contiguous K, matches B):
// lane m=L&15; lanes<16 hold K=0..15, lanes>=16 hold K=16..31
__device__ __forceinline__ v16bf load_A16x32(const __bf16* base, int stride,
                                             int m0, int kb, int lane) {
  int m  = m0 + (lane & 15);
  int hi = (lane >> 4) << 4;                 // 0 or 16
  const __bf16* p = base + m * stride + kb + hi;
  return *(const v16bf*)(p);                 // K = kb+hi .. +15
}

// B-matrix (32x16, bf16): lane n=L&15; lanes<16 hold K=0..15, lanes>=16 K=16..31
__device__ __forceinline__ v16bf load_B32x16(const __bf16* base, int stride,
                                             int row0, int kb, int lane) {
  int n  = lane & 15;
  int kh = (lane >> 4) << 4;                 // 0 or 16
  return *(const v16bf*)(base + (row0 + n) * stride + kb + kh);
}

// Sense-reversing grid barrier (48 persistent WGs).
__device__ __forceinline__ void grid_barrier(unsigned* cnt, unsigned* gen) {
  __syncthreads();
  if (threadIdx.x == 0) {
    unsigned g = __hip_atomic_load(gen, __ATOMIC_RELAXED, __HIP_MEMORY_SCOPE_AGENT);
    unsigned old = __hip_atomic_fetch_add(cnt, 1u, __ATOMIC_ACQ_REL, __HIP_MEMORY_SCOPE_AGENT);
    if (old == NWG - 1) {
      __hip_atomic_store(cnt, 0u, __ATOMIC_RELAXED, __HIP_MEMORY_SCOPE_AGENT);
      __hip_atomic_fetch_add(gen, 1u, __ATOMIC_RELEASE, __HIP_MEMORY_SCOPE_AGENT);
    } else {
      while (__hip_atomic_load(gen, __ATOMIC_ACQUIRE, __HIP_MEMORY_SCOPE_AGENT) == g)
        __builtin_amdgcn_s_sleep(2);
    }
  }
  __syncthreads();
}

__global__ void init_barrier(unsigned* bar) {
  if (threadIdx.x == 0) { bar[0] = 0u; bar[1] = 0u; }
}

// ---- persistent LSTM layer kernel ------------------------------------------

__global__ __launch_bounds__(THREADS, 1)
void lstm_layer_kernel(const float* __restrict__ x_in, int D,
                       const float* __restrict__ Wih,
                       const float* __restrict__ Whh,
                       const float* __restrict__ bih,
                       const float* __restrict__ bhh,
                       float* __restrict__ hs_out,
                       unsigned short* __restrict__ hg_raw,
                       unsigned* __restrict__ bar) {
  extern __shared__ __align__(16) char smem[];
  __bf16* whhL   = (__bf16*)smem;                 // [64][768]
  __bf16* wihL   = whhL + 64 * HLSTM;             // [64][D]
  __bf16* hbuf   = wihL + 64 * D;                 // [32][768]
  __bf16* xbuf   = hbuf + BATCH * HLSTM;          // [32][D]
  float*  gatesL = (float*)(xbuf + BATCH * D);    // [4][32][16]
  float*  cbuf   = gatesL + 4 * BATCH * 16;       // [32][16]
  float*  biasL  = cbuf + BATCH * 16;             // [64]

  const int tid    = threadIdx.x;
  const int lane   = tid & 31;
  const int wave   = tid >> 5;
  const int g      = wave >> 1;                   // gate 0..3 (i,f,g,o)
  const int m0     = (wave & 1) << 4;             // batch half 0 / 16
  const int slice0 = blockIdx.x * 16;             // owned hidden columns

  __bf16* hg = (__bf16*)hg_raw;

  // ---- one-time: convert weight slice to bf16 + fp32 bias sum in LDS ----
  for (int r = 0; r < 64; ++r) {
    int gg = r >> 4, j = r & 15;
    const float* sh = Whh + (size_t)(gg * HLSTM + slice0 + j) * HLSTM;
    for (int k = tid; k < HLSTM; k += THREADS) whhL[r * HLSTM + k] = to_bf16(sh[k]);
    const float* si = Wih + (size_t)(gg * HLSTM + slice0 + j) * D;
    for (int k = tid; k < D; k += THREADS)     wihL[r * D + k]     = to_bf16(si[k]);
  }
  if (tid < 64) {
    int gg = tid >> 4, j = tid & 15;
    biasL[tid] = bih[gg * HLSTM + slice0 + j] + bhh[gg * HLSTM + slice0 + j];
  }
  __syncthreads();

  // ---- time scan ----
  for (int t = 0; t < TSTEPS; ++t) {
    // stage x_t -> bf16 LDS
    const float4* xs = (const float4*)(x_in + (size_t)t * BATCH * D);
    const int nx4 = (BATCH * D) >> 2;
    for (int i = tid; i < nx4; i += THREADS) {
      float4 v = xs[i];
      int o = i << 2;
      xbuf[o]   = to_bf16(v.x); xbuf[o+1] = to_bf16(v.y);
      xbuf[o+2] = to_bf16(v.z); xbuf[o+3] = to_bf16(v.w);
    }
    if (t > 0) {
      // stage h_{t-1} (bf16 in global) -> LDS via async DMA (ASYNCcnt)
      const uint4* hs4 = (const uint4*)(hg + ((t + 1) & 1) * (BATCH * HLSTM));
      uint4* hb4 = (uint4*)hbuf;
      for (int i = tid; i < (BATCH * HLSTM) / 8; i += THREADS) {
        unsigned          ldsa = (unsigned)(unsigned long long)(hb4 + i);
        unsigned long long ga  = (unsigned long long)(hs4 + i);
        asm volatile("global_load_async_to_lds_b128 %0, %1, off"
                     :: "v"(ldsa), "v"(ga) : "memory");
      }
      asm volatile("s_wait_asynccnt 0x0" ::: "memory");
    }
    // prefetch next timestep's x slab while we compute (global_prefetch_b8)
    if (t + 1 < TSTEPS) {
      const char* nx = (const char*)(x_in + (size_t)(t + 1) * BATCH * D);
      const int nlines = (BATCH * D * 4) >> 8;     // 256B granules
      for (int i = tid; i < nlines; i += THREADS)
        __builtin_prefetch(nx + (i << 8), 0, 1);
    }
    __syncthreads();

    // gates tile = bias + x_t @ Wih^T (+ h_{t-1} @ Whh^T)
    float bn = biasL[(g << 4) + (lane & 15)];
    v8f acc = {bn, bn, bn, bn, bn, bn, bn, bn};
    for (int kb = 0; kb < D; kb += 32) {
      v16bf a = load_A16x32(xbuf, D, m0, kb, lane);
      v16bf b = load_B32x16(wihL, D, g << 4, kb, lane);
      acc = __builtin_amdgcn_wmma_f32_16x16x32_bf16(false, a, false, b,
                                                    (short)0, acc, false, false);
    }
    if (t > 0) {
      for (int kb = 0; kb < HLSTM; kb += 32) {
        v16bf a = load_A16x32(hbuf, HLSTM, m0, kb, lane);
        v16bf b = load_B32x16(whhL, HLSTM, g << 4, kb, lane);
        acc = __builtin_amdgcn_wmma_f32_16x16x32_bf16(false, a, false, b,
                                                      (short)0, acc, false, false);
      }
    }

    // spill accumulator tile (C layout: VGPR r -> M = r + 8*(lane>=16), N = lane&15)
    {
      int hi = (lane >> 4) << 3;
      int n  = lane & 15;
      for (int r = 0; r < 8; ++r)
        gatesL[(g << 9) + ((m0 + r + hi) << 4) + n] = acc[r];
    }
    __syncthreads();

    // elementwise LSTM update for our 32x16 slice
    __bf16* hgOut = hg + (t & 1) * (BATCH * HLSTM);
    for (int e = tid; e < BATCH * 16; e += THREADS) {
      int m = e >> 4, n = e & 15;
      float iv = sigmoidf_(gatesL[e]);
      float fv = sigmoidf_(gatesL[512 + e]);
      float gv = tanhf(gatesL[1024 + e]);
      float ov = sigmoidf_(gatesL[1536 + e]);
      float c  = (t > 0) ? cbuf[e] : 0.f;
      c = fv * c + iv * gv;
      cbuf[e] = c;
      float h = ov * tanhf(c);
      hs_out[((size_t)t * BATCH + m) * HLSTM + slice0 + n] = h;
      hgOut[m * HLSTM + slice0 + n] = to_bf16(h);
    }
    __threadfence();
    grid_barrier(bar, bar + 1);
  }
}

// ---- projection + log_softmax (wave per row, 41 classes) -------------------

__global__ __launch_bounds__(256)
void proj_kernel(const float* __restrict__ hs, const float* __restrict__ Wp,
                 const float* __restrict__ bp, float* __restrict__ out) {
  __shared__ float lg[8][48];
  const int wave = threadIdx.x >> 5;
  const int lane = threadIdx.x & 31;
  const int row  = blockIdx.x * 8 + wave;          // [0, T*B)
  const float* h = hs + (size_t)row * HLSTM;
  float* mylg = lg[wave];

  for (int v = 0; v < 41; ++v) {
    const float* w = Wp + v * HLSTM;
    float s = 0.f;
    for (int k = lane; k < HLSTM; k += 32) s += h[k] * w[k];
    for (int off = 16; off; off >>= 1) s += __shfl_xor(s, off);
    if (lane == 0) mylg[v] = s + bp[v];
  }
  // wave-internal LDS ops are in-order; safe to read back
  float m = -3.4e38f;
  for (int v = lane; v < 41; v += 32) m = fmaxf(m, mylg[v]);
  for (int off = 16; off; off >>= 1) m = fmaxf(m, __shfl_xor(m, off));
  float se = 0.f;
  for (int v = lane; v < 41; v += 32) se += __expf(mylg[v] - m);
  for (int off = 16; off; off >>= 1) se += __shfl_xor(se, off);
  float lse = m + __logf(se);
  for (int v = lane; v < 41; v += 32) out[(size_t)row * 41 + v] = mylg[v] - lse;
}

// ---- host entry ------------------------------------------------------------

extern "C" void kernel_launch(void* const* d_in, const int* in_sizes, int n_in,
                              void* d_out, int out_size, void* d_ws, size_t ws_size,
                              hipStream_t stream) {
  const float* x    = (const float*)d_in[0];
  const float* Wih0 = (const float*)d_in[1];
  const float* Wih  = (const float*)d_in[2];   // [4,3072,768]
  const float* Whh  = (const float*)d_in[3];   // [5,3072,768]
  const float* bih  = (const float*)d_in[4];   // [5,3072]
  const float* bhh  = (const float*)d_in[5];
  const float* Wp   = (const float*)d_in[6];
  const float* bp   = (const float*)d_in[7];
  float* out = (float*)d_out;

  char* ws = (char*)d_ws;
  const size_t HS_BYTES = (size_t)TSTEPS * BATCH * HLSTM * 4;   // 96 MB
  float*          hsA = (float*)(ws);
  float*          hsB = (float*)(ws + HS_BYTES);
  unsigned short* hg  = (unsigned short*)(ws + 2 * HS_BYTES);   // 2x[32][768] bf16
  unsigned*       bar = (unsigned*)(ws + 2 * HS_BYTES + 131072);

  init_barrier<<<dim3(1), dim3(64), 0, stream>>>(bar);

  auto layer = [&](const float* xin, int D, const float* wi, const float* wh,
                   const float* bi, const float* bh, float* hout) {
    size_t lds = (size_t)(64 * HLSTM + 64 * D + BATCH * HLSTM + BATCH * D) * 2
               + (size_t)(4 * BATCH * 16 + BATCH * 16 + 64) * 4;
    lstm_layer_kernel<<<dim3(NWG), dim3(THREADS), lds, stream>>>(
        xin, D, wi, wh, bi, bh, hout, hg, bar);
  };

  const size_t WSZ = (size_t)4 * HLSTM * HLSTM;   // 3072*768
  layer(x,   512, Wih0,          Whh + 0 * WSZ, bih + 0 * 3072, bhh + 0 * 3072, hsA);
  layer(hsA, 768, Wih + 0 * WSZ, Whh + 1 * WSZ, bih + 1 * 3072, bhh + 1 * 3072, hsB);
  layer(hsB, 768, Wih + 1 * WSZ, Whh + 2 * WSZ, bih + 2 * 3072, bhh + 2 * 3072, hsA);
  layer(hsA, 768, Wih + 2 * WSZ, Whh + 3 * WSZ, bih + 3 * 3072, bhh + 3 * 3072, hsB);
  layer(hsB, 768, Wih + 3 * WSZ, Whh + 4 * WSZ, bih + 4 * 3072, bhh + 4 * 3072, hsA);

  proj_kernel<<<dim3(TSTEPS * BATCH / 8), dim3(256), 0, stream>>>(hsA, Wp, bp, out);
}
